// MultiLoRADownsampleExtractor_15187004358684
// MI455X (gfx1250) — matrix operand order repaired and checked by reading, not compiled, not run on hardware
//
#include <hip/hip_runtime.h>

// ---------------- problem constants (from setup_inputs) ----------------
#define BB 4
#define LL 18
#define TT 512
#define DIN 2048
#define GG 6
#define NN 8
#define RR 16
#define DD 512
#define OUTD 1024
#define HH 8
#define HD 64
#define GROUP_SIZE 3

typedef __attribute__((ext_vector_type(16))) __bf16 v16bf;
typedef __attribute__((ext_vector_type(8)))  float  v8f;

union AFrag { v16bf v; unsigned u[8]; };
union BFrag { v16bf v; uint4 q[2]; };

// round-to-nearest-even fp32 -> bf16 bits
__device__ __forceinline__ unsigned short f2bf_rne(float f) {
    unsigned u = __float_as_uint(f);
    unsigned r = u + 0x7FFFu + ((u >> 16) & 1u);
    return (unsigned short)(r >> 16);
}
__device__ __forceinline__ float bf2f(unsigned short u) {
    return __uint_as_float(((unsigned)u) << 16);
}
// pack two floats into one dword of bf16 (truncation) via v_perm_b32
__device__ __forceinline__ unsigned pk2(float lo, float hi) {
    return __builtin_amdgcn_perm(__float_as_uint(hi), __float_as_uint(lo), 0x07060302u);
}

// ---------------------------------------------------------------------
// Fold LoRA into weights: for each (b,g) and the batch's one-hot expert n_b,
//   Weff_T[b*G+g][n][k] = bf16( W[g][k][n] + sum_r Bm[g,n_b,n,r]*A[g,n_b,r,k] )
// Stored TRANSPOSED (Dout major) so WMMA B-fragments are contiguous-K loads.
// ---------------------------------------------------------------------
__global__ __launch_bounds__(256) void fold_weights(
    const float* __restrict__ W,   // (G, Din, Dout)
    const float* __restrict__ A,   // (G, N, R, Din)
    const float* __restrict__ Bm,  // (G, N, Dout, R)
    const float* __restrict__ gate,// (B, N) one-hot
    unsigned short* __restrict__ wT, // (B*G, Dout, Din) bf16
    int Din, int Dout)
{
    int bg = blockIdx.y;
    int b = bg / GG, g = bg % GG;
    int nb = 0;
    #pragma unroll
    for (int n = 0; n < NN; ++n) if (gate[b * NN + n] > 0.5f) nb = n;

    long long idx = (long long)blockIdx.x * 256 + threadIdx.x;
    long long tot = (long long)Din * Dout;
    if (idx >= tot) return;
    int k = (int)(idx % Din);
    int n = (int)(idx / Din);

    const float* Abase = A + (((size_t)g * NN + nb) * RR) * Din;          // (R, Din)
    const float* Brow  = Bm + (((size_t)g * NN + nb) * Dout + n) * RR;    // (R,)
    float acc = W[((size_t)g * Din + k) * Dout + n];
    #pragma unroll
    for (int r = 0; r < RR; ++r)
        acc += Brow[r] * Abase[(size_t)r * Din + k];

    wT[((size_t)bg * Dout + n) * Din + k] = f2bf_rne(acc);
}

// ---------------------------------------------------------------------
// Big projection GEMM (Kp / Vp):  Out[bl][t][n] = x[bl][t][:] . WeffT[bg][n][:]
// bf16 WMMA 16x16x32. Block = 8 waves -> 64M x 128N C tile; each wave owns a
// 16M x 64N strip = 4 accumulators, so one A fragment (the expensive fp32->bf16
// packed operand) feeds 4 WMMAs per K step.
// ---------------------------------------------------------------------
__global__ __launch_bounds__(256) void gemm_proj_wmma(
    const float* __restrict__ x,          // (B*L, T, Din) fp32
    const unsigned short* __restrict__ wT,// (B*G, DD, Din) bf16 (transposed)
    const float* __restrict__ bias,       // (G, DD)
    unsigned short* __restrict__ out)     // (B*L, T, DD) bf16
{
    const int bl = blockIdx.z;
    const int b  = bl / LL;
    const int l  = bl % LL;
    const int g  = l / GROUP_SIZE;

    const int wave = threadIdx.x >> 5;
    const int lane = threadIdx.x & 31;
    const int half = lane >> 4;      // 0: lanes 0-15, 1: lanes 16-31
    const int lrow = lane & 15;

    const int m0 = blockIdx.x * 64  + (wave >> 1) * 16;
    const int n0 = blockIdx.y * 128 + (wave & 1) * 64;

    const float* xrow = x + (size_t)bl * TT * DIN + (size_t)(m0 + lrow) * DIN;
    // base of this lane's B column for sub-tile 0; sub s adds s*16 columns
    const unsigned short* bcol = wT + ((size_t)(b * GG + g) * DD + (n0 + lrow)) * DIN;
    const size_t subStride = (size_t)16 * DIN;

    const int ka0 = half * 16;   // A K-chunk base within the 32-step
    const int kb0 = half * 16;   // B K-chunk base within the 32-step

    v8f c0 = {}, c1 = {}, c2 = {}, c3 = {};
    #pragma unroll 2
    for (int kk = 0; kk < DIN; kk += 32) {
        // --- A fragment: 16 bf16 per lane = K {ka0..ka0+15}
        const float* pa = xrow + kk + ka0;
        __builtin_prefetch((const void*)(pa + 128), 0, 1);  // next K-tile of activations
        float4 f0 = *(const float4*)(pa);
        float4 f1 = *(const float4*)(pa + 4);
        float4 f2 = *(const float4*)(pa + 8);
        float4 f3 = *(const float4*)(pa + 12);
        AFrag a;
        a.u[0] = pk2(f0.x, f0.y); a.u[1] = pk2(f0.z, f0.w);
        a.u[2] = pk2(f1.x, f1.y); a.u[3] = pk2(f1.z, f1.w);
        a.u[4] = pk2(f2.x, f2.y); a.u[5] = pk2(f2.z, f2.w);
        a.u[6] = pk2(f3.x, f3.y); a.u[7] = pk2(f3.z, f3.w);

        // --- 4 B fragments: 16 contiguous bf16 of column n, K = kb0..kb0+15
        const unsigned short* pb = bcol + kk + kb0;
        BFrag b0, b1, b2, b3;
        b0.q[0] = *(const uint4*)(pb + 0 * subStride);
        b0.q[1] = *(const uint4*)(pb + 0 * subStride + 8);
        b1.q[0] = *(const uint4*)(pb + 1 * subStride);
        b1.q[1] = *(const uint4*)(pb + 1 * subStride + 8);
        b2.q[0] = *(const uint4*)(pb + 2 * subStride);
        b2.q[1] = *(const uint4*)(pb + 2 * subStride + 8);
        b3.q[0] = *(const uint4*)(pb + 3 * subStride);
        b3.q[1] = *(const uint4*)(pb + 3 * subStride + 8);

        c0 = __builtin_amdgcn_wmma_f32_16x16x32_bf16(false, a.v, false, b0.v, (short)0, c0, false, false);
        c1 = __builtin_amdgcn_wmma_f32_16x16x32_bf16(false, a.v, false, b1.v, (short)0, c1, false, false);
        c2 = __builtin_amdgcn_wmma_f32_16x16x32_bf16(false, a.v, false, b2.v, (short)0, c2, false, false);
        c3 = __builtin_amdgcn_wmma_f32_16x16x32_bf16(false, a.v, false, b3.v, (short)0, c3, false, false);
    }

    // epilogue: C element j -> row m0 + j + half*8, col = sub-tile col
    unsigned short* obl = out + (size_t)bl * TT * DD;
    #pragma unroll
    for (int s = 0; s < 4; ++s) {
        const v8f& c = (s == 0) ? c0 : (s == 1) ? c1 : (s == 2) ? c2 : c3;
        const int ncol = n0 + s * 16 + lrow;
        const float bb = bias[(size_t)g * DD + ncol];
        unsigned short* obase = obl + ncol;
        #pragma unroll
        for (int j = 0; j < 8; ++j) {
            int row = m0 + j + half * 8;
            obase[(size_t)row * DD] = f2bf_rne(c[j] + bb);
        }
    }
}

// ---------------------------------------------------------------------
// Qp GEMV: nq==1, Qp[bl][n] = qparams[l][:] . WeffT_q[bg][n][:] + bq[g][n]
// ---------------------------------------------------------------------
__global__ __launch_bounds__(256) void qp_gemv(
    const float* __restrict__ qp,          // (L, 1, Din)
    const unsigned short* __restrict__ wT, // (B*G, DD, Din) bf16
    const float* __restrict__ bias,        // (G, DD)
    float* __restrict__ Qp)                // (B*L, DD)
{
    __shared__ float xr[DIN];
    const int bl = blockIdx.x;
    const int b  = bl / LL;
    const int l  = bl % LL;
    const int g  = l / GROUP_SIZE;

    for (int k = threadIdx.x; k < DIN; k += 256)
        xr[k] = qp[(size_t)l * DIN + k];
    __syncthreads();

    const unsigned short* w = wT + (size_t)(b * GG + g) * DD * DIN;
    for (int n = threadIdx.x; n < DD; n += 256) {
        const unsigned short* wr = w + (size_t)n * DIN;
        float acc = bias[(size_t)g * DD + n];
        for (int k = 0; k < DIN; ++k)
            acc += xr[k] * bf2f(wr[k]);
        Qp[(size_t)bl * DD + n] = acc;
    }
}

// ---------------------------------------------------------------------
// Attention: one block per (bl, h). Single query, softmax over T=512,
// pooled written head_dim-major: pooled[bl][d*H + h].
// ---------------------------------------------------------------------
__global__ __launch_bounds__(256) void attn_pool(
    const float* __restrict__ Qp,           // (B*L, DD)
    const unsigned short* __restrict__ Kp,  // (B*L, T, DD) bf16
    const unsigned short* __restrict__ Vp,  // (B*L, T, DD) bf16
    float* __restrict__ pooled)             // (B*L, DD)
{
    __shared__ float q[HD];
    __shared__ float sc[TT];
    __shared__ float red[256];
    __shared__ float part[256];

    const int bl = blockIdx.x / HH;
    const int h  = blockIdx.x % HH;
    const int tid = threadIdx.x;

    if (tid < HD) q[tid] = Qp[(size_t)bl * DD + h * HD + tid];
    __syncthreads();

    const unsigned short* kb = Kp + (size_t)bl * TT * DD + h * HD;
    #pragma unroll
    for (int i = 0; i < 2; ++i) {
        int t = tid + i * 256;
        const unsigned short* kr = kb + (size_t)t * DD;
        float s = 0.f;
        for (int d = 0; d < HD; ++d) s += q[d] * bf2f(kr[d]);
        sc[t] = s * 0.125f;                 // 1/sqrt(64)
    }
    __syncthreads();

    // softmax max
    red[tid] = fmaxf(sc[tid], sc[tid + 256]);
    __syncthreads();
    for (int s = 128; s > 0; s >>= 1) {
        if (tid < s) red[tid] = fmaxf(red[tid], red[tid + s]);
        __syncthreads();
    }
    const float mx = red[0];
    __syncthreads();

    float e0 = __expf(sc[tid] - mx);
    float e1 = __expf(sc[tid + 256] - mx);
    sc[tid] = e0; sc[tid + 256] = e1;
    red[tid] = e0 + e1;
    __syncthreads();
    for (int s = 128; s > 0; s >>= 1) {
        if (tid < s) red[tid] += red[tid + s];
        __syncthreads();
    }
    const float inv = 1.f / red[0];
    __syncthreads();

    // pooled[d] = sum_t p[t] * V[t][d]; 4 T-chunks x 64 d
    const int d = tid & 63;
    const int ch = tid >> 6;
    const unsigned short* vb = Vp + (size_t)bl * TT * DD + h * HD + d;
    float acc = 0.f;
    for (int t = ch * 128; t < ch * 128 + 128; ++t)
        acc += sc[t] * bf2f(vb[(size_t)t * DD]);
    part[tid] = acc;
    __syncthreads();
    if (tid < HD) {
        float s = part[tid] + part[tid + 64] + part[tid + 128] + part[tid + 192];
        pooled[(size_t)bl * DD + tid * HH + h] = s * inv;   // head_dim-major
    }
}

// ---------------------------------------------------------------------
// Output GEMV: feat[bl][o] = pooled[bl][:] . WeffT_o[bg][o][:] + bo[g][o]
// ---------------------------------------------------------------------
__global__ __launch_bounds__(256) void out_gemv(
    const float* __restrict__ pooled,       // (B*L, DD)
    const unsigned short* __restrict__ wT,  // (B*G, OUTD, DD) bf16
    const float* __restrict__ bias,         // (G, OUTD)
    float* __restrict__ out)                // (B*L, OUTD)
{
    __shared__ float pd[DD];
    const int bl = blockIdx.x;
    const int b  = bl / LL;
    const int l  = bl % LL;
    const int g  = l / GROUP_SIZE;

    for (int d = threadIdx.x; d < DD; d += 256)
        pd[d] = pooled[(size_t)bl * DD + d];
    __syncthreads();

    const unsigned short* w = wT + (size_t)(b * GG + g) * OUTD * DD;
    for (int o = threadIdx.x; o < OUTD; o += 256) {
        const unsigned short* wr = w + (size_t)o * DD;
        float acc = bias[(size_t)g * OUTD + o];
        for (int d = 0; d < DD; ++d)
            acc += pd[d] * bf2f(wr[d]);
        out[(size_t)bl * OUTD + o] = acc;
    }
}

// ---------------------------------------------------------------------
extern "C" void kernel_launch(void* const* d_in, const int* in_sizes, int n_in,
                              void* d_out, int out_size, void* d_ws, size_t ws_size,
                              hipStream_t stream) {
    (void)in_sizes; (void)n_in; (void)out_size; (void)ws_size;
    const float* K   = (const float*)d_in[0];
    const float* V   = (const float*)d_in[1];
    const float* gate= (const float*)d_in[2];
    const float* qpar= (const float*)d_in[3];
    const float* Wq = (const float*)d_in[4];  const float* bq = (const float*)d_in[5];
    const float* Aq = (const float*)d_in[6];  const float* Bq = (const float*)d_in[7];
    const float* Wk = (const float*)d_in[8];  const float* bk = (const float*)d_in[9];
    const float* Ak = (const float*)d_in[10]; const float* Bk = (const float*)d_in[11];
    const float* Wv = (const float*)d_in[12]; const float* bv = (const float*)d_in[13];
    const float* Av = (const float*)d_in[14]; const float* Bv = (const float*)d_in[15];
    const float* Wo = (const float*)d_in[16]; const float* bo = (const float*)d_in[17];
    const float* Ao = (const float*)d_in[18]; const float* Bo = (const float*)d_in[19];

    // workspace carve-up (256B aligned)
    char* ws = (char*)d_ws;
    size_t off = 0;
    auto carve = [&](size_t bytes) { char* p = ws + off; off += (bytes + 255) & ~(size_t)255; return p; };
    const size_t BG = (size_t)BB * GG, BL = (size_t)BB * LL;
    unsigned short* wqT = (unsigned short*)carve(BG * DD * DIN * 2);
    unsigned short* wkT = (unsigned short*)carve(BG * DD * DIN * 2);
    unsigned short* wvT = (unsigned short*)carve(BG * DD * DIN * 2);
    unsigned short* woT = (unsigned short*)carve(BG * OUTD * DD * 2);
    unsigned short* Kp  = (unsigned short*)carve(BL * TT * DD * 2);
    unsigned short* Vp  = (unsigned short*)carve(BL * TT * DD * 2);
    float* Qp     = (float*)carve(BL * DD * 4);
    float* pooled = (float*)carve(BL * DD * 4);

    // 1) fold LoRA into per-(b,g) weights
    {
        dim3 gbig(((size_t)DIN * DD + 255) / 256, (unsigned)BG);
        fold_weights<<<gbig, 256, 0, stream>>>(Wq, Aq, Bq, gate, wqT, DIN, DD);
        fold_weights<<<gbig, 256, 0, stream>>>(Wk, Ak, Bk, gate, wkT, DIN, DD);
        fold_weights<<<gbig, 256, 0, stream>>>(Wv, Av, Bv, gate, wvT, DIN, DD);
        dim3 go(((size_t)DD * OUTD + 255) / 256, (unsigned)BG);
        fold_weights<<<go, 256, 0, stream>>>(Wo, Ao, Bo, gate, woT, DD, OUTD);
    }

    // 2) Q projection (M=1 GEMV)
    qp_gemv<<<(unsigned)BL, 256, 0, stream>>>(qpar, wqT, bq, Qp);

    // 3) K/V projections via bf16 WMMA (64x128 block tile, 16x64 wave tile)
    {
        dim3 gg(TT / 64, DD / 128, (unsigned)BL);
        gemm_proj_wmma<<<gg, 256, 0, stream>>>(K, wkT, bk, Kp);
        gemm_proj_wmma<<<gg, 256, 0, stream>>>(V, wvT, bv, Vp);
    }

    // 4) single-query attention + head_dim-major pooling
    attn_pool<<<(unsigned)(BL * HH), 256, 0, stream>>>(Qp, Kp, Vp, pooled);

    // 5) output projection (M=1 GEMV)
    out_gemv<<<(unsigned)BL, 256, 0, stream>>>(pooled, woT, bo, (float*)d_out);
}
